// BottleNeckKAGNConvNDLayer_60885456388455
// MI455X (gfx1250) — compile-verified, matrix operand order checked by reading, not run on hardware
//
#include <hip/hip_runtime.h>
#include <hip/hip_bf16.h>
#include <cstdint>
#include <cstddef>

// ---------- types ----------
typedef __bf16 bf16_t;
typedef bf16_t v16bf __attribute__((ext_vector_type(16)));
typedef float  v8f   __attribute__((ext_vector_type(8)));
typedef unsigned int   u32;
typedef unsigned short u16;

union Frag { v16bf v; u32 u[8]; };

__device__ __forceinline__ u16 f2bf(float f) {
  bf16_t h = (bf16_t)f;
  return __builtin_bit_cast(u16, h);
}
__device__ __forceinline__ u32 pack2bf(float a, float b) {
  return (u32)f2bf(a) | ((u32)f2bf(b) << 16);
}
__device__ __forceinline__ float bf2f(u16 u) {
  bf16_t h = __builtin_bit_cast(bf16_t, u);
  return (float)h;
}
__device__ __forceinline__ float silu_f(float x) { return x / (1.0f + __expf(-x)); }

// ---------- problem dims ----------
#define BSZ   8
#define HDIM  64
#define WDIM  64
#define NPIX  32768          // BSZ*HDIM*WDIM
#define CIN   512
#define INNER 128
#define PASS_ELEMS 16777216u // NPIX*CIN

// ---------- fp32 -> bf16 convert (8 elems / thread, b128 in+out) ----------
__global__ void k_cvt_bf16(const float* __restrict__ x, u32* __restrict__ xb, unsigned n8) {
  for (unsigned i = blockIdx.x * 256u + threadIdx.x; i < n8; i += gridDim.x * 256u) {
    const float4* p = (const float4*)x + (size_t)i * 2;
    float4 f0 = p[0], f1 = p[1];
    uint4 o;
    o.x = pack2bf(f0.x, f0.y); o.y = pack2bf(f0.z, f0.w);
    o.z = pack2bf(f1.x, f1.y); o.w = pack2bf(f1.z, f1.w);
    ((uint4*)xb)[i] = o;
  }
}

// ---------- weight pack: W[n][k] (strided fp32) -> WMMA-B fragment-ready bf16 pairs ----------
__global__ void k_pack_w(const float* __restrict__ W, u32* __restrict__ P,
                         int N, int K, int sn, int sk, int off) {
  int total = (N * K) >> 1;
  int NT16 = N >> 4;
  for (int u = blockIdx.x * 256 + threadIdx.x; u < total; u += gridDim.x * 256) {
    int j = u & 7, lane = (u >> 3) & 31, tile = u >> 8;
    int nt = tile % NT16, kcI = tile / NT16;
    int kk = kcI * 32 + ((lane >> 4) << 3) + ((j & 3) << 1) + ((j >> 2) << 4);
    int nn = (nt << 4) + (lane & 15);
    float w0 = W[(size_t)nn * sn + (size_t)kk * sk + off];
    float w1 = W[(size_t)nn * sn + (size_t)(kk + 1) * sk + off];
    P[u] = pack2bf(w0, w1);
  }
}

// ---------- depthwise 3x3 with fused silu (channel-last, 2 channels / thread) ----------
__global__ void k_dwconv(const u32* __restrict__ xb2, const float* __restrict__ dww,
                         u32* __restrict__ db2) {
  size_t idx = (size_t)blockIdx.x * 256 + threadIdx.x;   // over PASS_ELEMS/2 bf16-pairs
  if (idx >= (PASS_ELEMS >> 1)) return;
  int cp = (int)(idx & 255);
  size_t pix = idx >> 8;
  int w = (int)(pix & 63), h = (int)((pix >> 6) & 63);
  size_t n = pix >> 12;
  int c0 = cp << 1;
  float a0 = 0.0f, a1 = 0.0f;
  #pragma unroll
  for (int dy = -1; dy <= 1; ++dy) {
    int hh = h + dy; if (hh < 0 || hh > 63) continue;
    #pragma unroll
    for (int dx = -1; dx <= 1; ++dx) {
      int ww = w + dx; if (ww < 0 || ww > 63) continue;
      u32 pr = xb2[((n * 4096) + (size_t)(hh * 64 + ww)) * 256 + cp];
      float v0 = bf2f((u16)(pr & 0xffff));
      float v1 = bf2f((u16)(pr >> 16));
      int tap = (dy + 1) * 3 + (dx + 1);
      a0 += silu_f(v0) * dww[c0 * 9 + tap];
      a1 += silu_f(v1) * dww[(c0 + 1) * 9 + tap];
    }
  }
  db2[idx] = pack2bf(a0, a1);
}

// ---------- one k-step load: 2 A subtiles + 4 B tiles (12 x b128); advances pointers ----------
// Loop-carried pointer updates bound scheduler lookahead to <= 2 steps (no spill blowup).
template<int NT16, bool MASKA>
__device__ __forceinline__ void loadstep(const u32*& a0p, const u32*& a1p, const u32*& bpc,
                                         bool v0, bool v1, Frag (&af)[2], Frag (&bf)[4]) {
  *(uint4*)&af[0].u[0] = *(const uint4*)(a0p);
  *(uint4*)&af[0].u[4] = *(const uint4*)(a0p + 8);
  *(uint4*)&af[1].u[0] = *(const uint4*)(a1p);
  *(uint4*)&af[1].u[4] = *(const uint4*)(a1p + 8);
  if (MASKA) {
    if (!v0) {
      #pragma unroll
      for (int q = 0; q < 8; ++q) af[0].u[q] = 0u;
    }
    if (!v1) {
      #pragma unroll
      for (int q = 0; q < 8; ++q) af[1].u[q] = 0u;
    }
  }
  #pragma unroll
  for (int t = 0; t < 4; ++t) {
    *(uint4*)&bf[t].u[0] = *(const uint4*)(bpc + t * 256);
    *(uint4*)&bf[t].u[4] = *(const uint4*)(bpc + t * 256 + 4);
  }
  a0p += 16; a1p += 16; bpc += NT16 * 256;
  __builtin_amdgcn_sched_group_barrier(0x020, 12, 0);   // pin this step's 12 VMEM reads
}

// ---------- 8 WMMAs: 2 M-subtiles x 4 N-tiles sharing B fragments ----------
__device__ __forceinline__ void wmma8(const Frag (&a)[2], const Frag (&b)[4], v8f (*acc)[4]) {
  #pragma unroll
  for (int t = 0; t < 4; ++t)
    acc[0][t] = __builtin_amdgcn_wmma_f32_16x16x32_bf16(false, a[0].v, false, b[t].v,
                                                        (short)0, acc[0][t], false, false);
  #pragma unroll
  for (int t = 0; t < 4; ++t)
    acc[1][t] = __builtin_amdgcn_wmma_f32_16x16x32_bf16(false, a[1].v, false, b[t].v,
                                                        (short)0, acc[1][t], false, false);
  __builtin_amdgcn_sched_group_barrier(0x008, 8, 0);    // pin this step's 8 WMMAs
}

// ---------- software-pipelined wave GEMM: M=32 x N=64 per wave ----------
// Runtime loop (no full unroll) + ping-pong fragment buffers: loads of step s+1
// issue under the 8-deep wmma group of step s; pointer chain bounds lookahead.
template<int KTOT, int NT16, bool MASKA>
__device__ __forceinline__ void gemm_waveloop2(const u32* __restrict__ ap0,
                                               const u32* __restrict__ ap1,
                                               const u32* __restrict__ bp,
                                               bool v0, bool v1, v8f (*acc)[4]) {
  constexpr int NI = KTOT / 32;
  static_assert((NI & 1) == 0, "even k-step count required");
  const u32 *a0p = ap0, *a1p = ap1, *bpc = bp;
  Frag aA[2], bA[4], aB[2], bB[4];
  loadstep<NT16, MASKA>(a0p, a1p, bpc, v0, v1, aA, bA);
  #pragma unroll 1
  for (int i = 0; i < NI / 2 - 1; ++i) {
    loadstep<NT16, MASKA>(a0p, a1p, bpc, v0, v1, aB, bB);
    wmma8(aA, bA, acc);
    loadstep<NT16, MASKA>(a0p, a1p, bpc, v0, v1, aA, bA);
    wmma8(aB, bB, acc);
  }
  loadstep<NT16, MASKA>(a0p, a1p, bpc, v0, v1, aB, bB);
  wmma8(aA, bA, acc);
  wmma8(aB, bB, acc);
}

// ---------- GEMM: basis = db @ pw^T  (M=32768,N=512,K=512), store f32 ----------
__global__ void __launch_bounds__(256, 1)
k_gemm_pw(const u16* __restrict__ A, const u32* __restrict__ Bp, float* __restrict__ basis) {
  int lane = threadIdx.x & 31, wave = threadIdx.x >> 5;
  int m0 = blockIdx.x * 128 + (wave & 3) * 32;
  int nt0 = blockIdx.y * 8 + (wave >> 2) * 4;
  v8f acc[2][4] = {};
  const u32* ap0 = (const u32*)A + (size_t)(m0 + (lane & 15)) * 256 + ((lane >> 4) << 2);
  const u32* ap1 = ap0 + (size_t)16 * 256;
  const u32* bp = Bp + (size_t)nt0 * 256 + lane * 8;
  gemm_waveloop2<512, 32, false>(ap0, ap1, bp, true, true, acc);
  #pragma unroll
  for (int s = 0; s < 2; ++s)
    #pragma unroll
    for (int t = 0; t < 4; ++t) {
      int col = (nt0 + t) * 16 + (lane & 15);
      #pragma unroll
      for (int r = 0; r < 8; ++r) {
        size_t pix = (size_t)m0 + s * 16 + r + ((lane >> 4) << 3);
        basis[pix * 512 + col] = acc[s][t][r];
      }
    }
}

// ---------- GEMM: xi = tanh(x @ inner^T) -> Gram poly -> silu -> gb bf16 ----------
__global__ void __launch_bounds__(256, 1)
k_gemm_inner(const u16* __restrict__ A, const u32* __restrict__ Bp,
             const float* __restrict__ beta_w, u16* __restrict__ gb) {
  int lane = threadIdx.x & 31, wave = threadIdx.x >> 5;
  int m0 = blockIdx.x * 128 + (wave & 3) * 32;
  int nt0 = (wave >> 2) * 4;
  v8f acc[2][4] = {};
  const u32* ap0 = (const u32*)A + (size_t)(m0 + (lane & 15)) * 256 + ((lane >> 4) << 2);
  const u32* ap1 = ap0 + (size_t)16 * 256;
  const u32* bp = Bp + (size_t)nt0 * 256 + lane * 8;
  gemm_waveloop2<512, 8, false>(ap0, ap1, bp, true, true, acc);
  float bt2 = 2.25f * beta_w[1];                  // n=1,m=2: 3*1*1 / (4/3)
  float bt3 = (20.0f * 15.0f / 9.0f) * beta_w[2]; // n=2,m=3: 5*1*4 / (9/15)
  const float s1 = 0.73105857863f;                // silu(1): constant degree-0 block
  #pragma unroll
  for (int s = 0; s < 2; ++s)
    #pragma unroll
    for (int t = 0; t < 4; ++t) {
      int ci = (nt0 + t) * 16 + (lane & 15);
      #pragma unroll
      for (int r = 0; r < 8; ++r) {
        size_t pix = (size_t)m0 + s * 16 + r + ((lane >> 4) << 3);
        float tt = tanhf(acc[s][t][r]);
        float p2 = tt * tt - bt2;
        float p3 = tt * p2 - bt3 * tt;
        u16* row = gb + pix * 512;
        row[ci]       = f2bf(s1);
        row[128 + ci] = f2bf(silu_f(tt));
        row[256 + ci] = f2bf(silu_f(p2));
        row[384 + ci] = f2bf(silu_f(p3));
      }
    }
}

// ---------- implicit 3x3 conv GEMM: block = 2 image rows (M=128), N=128, K=512*9 ----------
__global__ void __launch_bounds__(256, 1)
k_polyconv(const u16* __restrict__ gb, const u32* __restrict__ polyP, u16* __restrict__ yb) {
  int lane = threadIdx.x & 31, wave = threadIdx.x >> 5;
  int n = blockIdx.x >> 5;
  int h0 = (blockIdx.x & 31) << 1;
  int r = h0 + ((wave & 3) >> 1);        // this wave's image row (wave-uniform)
  int w0 = ((wave & 3) & 1) * 32;        // 32-pixel column strip
  int nt0 = (wave >> 2) * 4;
  int wl = w0 + (lane & 15);
  v8f acc[2][4] = {};
  for (int tap = 0; tap < 9; ++tap) {
    int dy = tap / 3 - 1, dx = tap % 3 - 1;
    int hh = r + dy;
    if (hh < 0 || hh > 63) continue;     // wave-uniform skip: EXEC stays all-ones
    int wwa = wl + dx, wwb = wl + 16 + dx;
    bool va = (wwa >= 0 && wwa <= 63);
    bool vb = (wwb >= 0 && wwb <= 63);
    int wca = min(max(wwa, 0), 63);      // clamp keeps addresses in-bounds
    int wcb = min(max(wwb, 0), 63);
    size_t rowbase = (size_t)n * 4096 + (size_t)hh * 64;
    const u32* ap0 = (const u32*)gb + (rowbase + wca) * 256 + ((lane >> 4) << 2);
    const u32* ap1 = (const u32*)gb + (rowbase + wcb) * 256 + ((lane >> 4) << 2);
    const u32* bp = polyP + tap * 32768 + (size_t)nt0 * 256 + lane * 8;
    gemm_waveloop2<512, 8, true>(ap0, ap1, bp, va, vb, acc);
  }
  #pragma unroll
  for (int s = 0; s < 2; ++s)
    #pragma unroll
    for (int t = 0; t < 4; ++t) {
      int col = (nt0 + t) * 16 + (lane & 15);
      #pragma unroll
      for (int rr = 0; rr < 8; ++rr) {
        int wp = w0 + s * 16 + rr + ((lane >> 4) << 3);
        size_t pix = (size_t)n * 4096 + (size_t)r * 64 + wp;
        yb[pix * 128 + col] = f2bf(acc[s][t][rr]);
      }
    }
}

// ---------- GEMM: z = yb @ out^T + basis (M=32768,N=512,K=128), store f32 ----------
__global__ void __launch_bounds__(256, 1)
k_gemm_out(const u16* __restrict__ A, const u32* __restrict__ Bp,
           const float* __restrict__ basis, float* __restrict__ z) {
  int lane = threadIdx.x & 31, wave = threadIdx.x >> 5;
  int m0 = blockIdx.x * 128 + (wave & 3) * 32;
  int nt0 = blockIdx.y * 8 + (wave >> 2) * 4;
  v8f acc[2][4] = {};
  const u32* ap0 = (const u32*)A + (size_t)(m0 + (lane & 15)) * 64 + ((lane >> 4) << 2);
  const u32* ap1 = ap0 + (size_t)16 * 64;
  const u32* bp = Bp + (size_t)nt0 * 256 + lane * 8;
  gemm_waveloop2<128, 32, false>(ap0, ap1, bp, true, true, acc);
  #pragma unroll
  for (int s = 0; s < 2; ++s)
    #pragma unroll
    for (int t = 0; t < 4; ++t) {
      int col = (nt0 + t) * 16 + (lane & 15);
      #pragma unroll
      for (int r = 0; r < 8; ++r) {
        size_t pix = (size_t)m0 + s * 16 + r + ((lane >> 4) << 3);
        size_t idx = pix * 512 + col;
        z[idx] = acc[s][t][r] + basis[idx];
      }
    }
}

// ---------- instance-norm stats, stage 1: coalesced partials per 64-pixel group ----------
__global__ void k_stats1(const float* __restrict__ z, float* __restrict__ psum,
                         float* __restrict__ psq) {
  int g = blockIdx.x;                 // 0..511 (64 pixels each)
  int c0 = threadIdx.x * 2;
  const float* zp = z + (size_t)g * 64 * 512 + c0;
  float s0 = 0, s1 = 0, q0 = 0, q1 = 0;
  for (int p = 0; p < 64; ++p) {
    float2 v = *(const float2*)(zp + (size_t)p * 512);
    s0 += v.x; s1 += v.y; q0 += v.x * v.x; q1 += v.y * v.y;
  }
  *(float2*)(psum + (size_t)g * 512 + c0) = make_float2(s0, s1);
  *(float2*)(psq  + (size_t)g * 512 + c0) = make_float2(q0, q1);
}

// ---------- instance-norm stats, stage 2: reduce 64 groups per (n,c) ----------
__global__ void k_stats2(const float* __restrict__ psum, const float* __restrict__ psq,
                         float* __restrict__ mu, float* __restrict__ rstd) {
  int n = blockIdx.x;                 // 0..7
  int c0 = threadIdx.x * 2;
  float s0 = 0, s1 = 0, q0 = 0, q1 = 0;
  for (int gg = 0; gg < 64; ++gg) {
    size_t base = ((size_t)n * 64 + gg) * 512 + c0;
    float2 a = *(const float2*)(psum + base);
    float2 b = *(const float2*)(psq + base);
    s0 += a.x; s1 += a.y; q0 += b.x; q1 += b.y;
  }
  const float inv = 1.0f / 4096.0f;
  float m0 = s0 * inv, m1 = s1 * inv;
  int b = n * 512 + c0;
  mu[b] = m0;     mu[b + 1] = m1;
  rstd[b]     = rsqrtf(q0 * inv - m0 * m0 + 1e-5f);
  rstd[b + 1] = rsqrtf(q1 * inv - m1 * m1 + 1e-5f);
}

// ---------- normalize + affine + silu -> fp32 output ----------
__global__ void k_final(const float* __restrict__ z, const float* __restrict__ mu,
                        const float* __restrict__ rstd, const float* __restrict__ gamma,
                        const float* __restrict__ beta, float* __restrict__ out) {
  for (size_t idx = (size_t)blockIdx.x * 256 + threadIdx.x; idx < (size_t)PASS_ELEMS;
       idx += (size_t)gridDim.x * 256) {
    int c = (int)(idx & 511);
    int n = (int)((idx >> 9) >> 12);
    int b = n * 512 + c;
    float v = (z[idx] - mu[b]) * rstd[b] * gamma[c] + beta[c];
    out[idx] = silu_f(v);
  }
}

// ---------- host ----------
extern "C" void kernel_launch(void* const* d_in, const int* in_sizes, int n_in,
                              void* d_out, int out_size, void* d_ws, size_t ws_size,
                              hipStream_t stream) {
  (void)in_sizes; (void)n_in; (void)out_size; (void)ws_size;
  const float* in_rgb   = (const float*)d_in[0];
  const float* in_ir    = (const float*)d_in[1];
  const float* dw_w     = (const float*)d_in[2]; // [512,9]
  const float* pw_w     = (const float*)d_in[3]; // [512,512]
  const float* inner_w  = (const float*)d_in[4]; // [128,512]
  const float* out_w    = (const float*)d_in[5]; // [512,128]
  const float* gamma    = (const float*)d_in[6]; // [512]
  const float* beta_p   = (const float*)d_in[7]; // [512]
  const float* poly_w   = (const float*)d_in[8]; // [128,512,3,3]
  const float* beta_w   = (const float*)d_in[9]; // [4]

  char* ws = (char*)d_ws;
  size_t off = 0;
  auto carve = [&](size_t bytes) { char* p = ws + off; off += (bytes + 255) & ~(size_t)255; return p; };
  u16*   xb    = (u16*)  carve((size_t)PASS_ELEMS * 2);        // x bf16
  u16*   db    = (u16*)  carve((size_t)PASS_ELEMS * 2);        // depthwise(silu(x)) bf16
  u16*   gb    = (u16*)  carve((size_t)PASS_ELEMS * 2);        // gram basis bf16 [pix][512]
  u16*   yb    = (u16*)  carve((size_t)NPIX * INNER * 2);      // poly conv out bf16
  float* basis = (float*)carve((size_t)PASS_ELEMS * 4);
  float* z     = (float*)carve((size_t)PASS_ELEMS * 4);
  u32*   pwP   = (u32*)  carve((size_t)512 * 512 * 2);
  u32*   inP   = (u32*)  carve((size_t)128 * 512 * 2);
  u32*   outP  = (u32*)  carve((size_t)512 * 128 * 2);
  u32*   polyP = (u32*)  carve((size_t)9 * 512 * 128 * 2);
  float* psum  = (float*)carve((size_t)512 * 512 * 4);
  float* psq   = (float*)carve((size_t)512 * 512 * 4);
  float* mu    = (float*)carve((size_t)BSZ * 512 * 4);
  float* rstd  = (float*)carve((size_t)BSZ * 512 * 4);

  // pack weights to fragment-ready bf16 (tiny; resident in L2 thereafter)
  k_pack_w<<<512, 256, 0, stream>>>(pw_w,    pwP,  512, 512, 512, 1, 0);
  k_pack_w<<<128, 256, 0, stream>>>(inner_w, inP,  128, 512, 512, 1, 0);
  k_pack_w<<<128, 256, 0, stream>>>(out_w,   outP, 512, 128, 128, 1, 0);
  for (int tap = 0; tap < 9; ++tap)
    k_pack_w<<<128, 256, 0, stream>>>(poly_w, polyP + tap * 32768, 128, 512, 512 * 9, 9, tap);

  for (int pass = 0; pass < 2; ++pass) {
    const float* x = pass ? in_ir : in_rgb;
    float* out = (float*)d_out + (size_t)pass * PASS_ELEMS;

    k_cvt_bf16 <<<4096, 256, 0, stream>>>(x, (u32*)xb, PASS_ELEMS / 8);
    k_dwconv   <<<PASS_ELEMS / 512, 256, 0, stream>>>((const u32*)xb, dw_w, (u32*)db);
    k_gemm_pw  <<<dim3(NPIX / 128, 4), 256, 0, stream>>>(db, pwP, basis);
    k_gemm_inner<<<dim3(NPIX / 128, 1), 256, 0, stream>>>(xb, inP, beta_w, gb);
    k_polyconv <<<BSZ * HDIM / 2, 256, 0, stream>>>(gb, polyP, yb);
    k_gemm_out <<<dim3(NPIX / 128, 4), 256, 0, stream>>>(yb, outP, basis, z);
    k_stats1   <<<512, 256, 0, stream>>>(z, psum, psq);
    k_stats2   <<<BSZ, 256, 0, stream>>>(psum, psq, mu, rstd);
    k_final    <<<8192, 256, 0, stream>>>(z, mu, rstd, gamma, beta_p, out);
  }
}
